// MixtralSparseMoeBlock_6073083756872
// MI455X (gfx1250) — matrix operand order repaired and checked by reading, not compiled, not run on hardware
//
#include <hip/hip_runtime.h>

#define H_DIM 1024
#define I_DIM 3584
#define N_EXP 8
#define M_TILE 32
#define IC 256          // I-chunk per iteration (16 waves x 16 cols)
#define TROW 266        // padded t row (bf16): 133-dword stride -> conflict-free A reads
#define NKB (IC / 32)   // k-blocks per chunk in GEMM2
#define SLOT_DW 544     // 32x16 fp32 tile + 16-row skew (+16 dwords) -> conflict-free

#if defined(__gfx1250__) && __has_builtin(__builtin_amdgcn_global_load_async_to_lds_b128)
#define HAVE_ASYNC 1
#else
#define HAVE_ASYNC 0
#endif

typedef int v4i __attribute__((vector_size(16)));
typedef __attribute__((address_space(1))) v4i* as1_v4i;
typedef __attribute__((address_space(3))) v4i* as3_v4i;

typedef __attribute__((ext_vector_type(16))) __bf16 v16bf;
typedef __attribute__((ext_vector_type(8)))  float  v8f;

union Frag16 { v16bf v; unsigned int d[8]; unsigned short h[16]; };
union FragAcc { v8f v; float f[8]; };

// Pure (non-volatile) so the scheduler can batch loads ahead of conversions.
__device__ __forceinline__ unsigned int cvtpk(float lo, float hi) {
  unsigned int r;
  asm("v_cvt_pk_bf16_f32 %0, %1, %2" : "=v"(r) : "v"(lo), "v"(hi));
  return r;
}
__device__ __forceinline__ unsigned short f2bf(float x) {
  unsigned int u = __float_as_uint(x);
  u += 0x7FFFu + ((u >> 16) & 1u);
  return (unsigned short)(u >> 16);
}

template <int N>
__device__ __forceinline__ void wait_async_c() {
#if HAVE_ASYNC
#if __has_builtin(__builtin_amdgcn_s_wait_asynccnt)
  __builtin_amdgcn_s_wait_asynccnt(N);
#else
  asm volatile("s_wait_asynccnt %0" ::"i"(N));
#endif
#endif
}

// Stage a 32x16 fp32 tile global -> LDS with +16-dword skew on rows 16..31.
// Async on gfx1250 (4 x b128 per wave), sync fallback otherwise.
__device__ __forceinline__ void stage_tile(const float* __restrict__ g, size_t rstride,
                                           unsigned int* lbase, int lane) {
  const int row = lane >> 2;        // 0..7
  const int cb  = (lane & 3) << 2;  // column offset in floats (16B per lane)
#pragma unroll
  for (int q = 0; q < 4; ++q) {
    const int rr = (q << 3) + row;  // 0..31
    const float* src = g + (size_t)rr * rstride + cb;
    unsigned int* dst = lbase + rr * 16 + ((rr >> 4) << 4) + cb;  // skewed, 16B-aligned
#if HAVE_ASYNC
    // global generic ptr bit-pattern == AS1; LDS offset == low 32 bits (ISA aperture rule)
    __builtin_amdgcn_global_load_async_to_lds_b128(
        (as1_v4i)(unsigned long long)src,
        (as3_v4i)(unsigned int)(unsigned long long)dst, 0, 0);
#else
    *(uint4*)dst = *(const uint4*)src;
#endif
  }
}

// Build a bf16 B fragment from a skewed staged fp32 32x16 tile in LDS.
// B layout: lanes 0-15 hold K 0-7,16-23, lanes 16-31 hold K 8-15,24-31, K-pairs per VGPR.
// Both lane-halves read rows k and k+8, on the same side of the skew (128 dwords apart).
__device__ __forceinline__ void build_fragB(Frag16& b, const unsigned int* slot,
                                            int lhalf, int l15) {
  const float* t = (const float*)slot;
  const int base = lhalf * 128 + l15;  // rowoff(k)=k*16+((k>>4)<<4); k=lhalf*8+2j (+16 for j>=4)
#pragma unroll
  for (int j = 0; j < 8; ++j) {
    const int o = base + ((j < 4) ? (32 * j) : (272 + 32 * (j - 4)));
    b.d[j] = cvtpk(t[o], t[o + 16]);
  }
}

__device__ __forceinline__ v8f wmma_bf16(const Frag16& a, const Frag16& b, v8f c) {
  return __builtin_amdgcn_wmma_f32_16x16x32_bf16(false, a.v, false, b.v, (short)0, c,
                                                 false, false);
}

// ---------------- Router: logits, top-2, softmax -> dense [T,8] weight table ----------
__global__ void moe_router_kernel(const float* __restrict__ hs,
                                  const float* __restrict__ gw,
                                  float* __restrict__ logits,
                                  float* __restrict__ wbuf) {
  __shared__ float lg[32 * 8];
  const int tid = threadIdx.x;
  const int token = blockIdx.x * 32 + (tid >> 3);
  const int e = tid & 7;
  const float* hp = hs + (size_t)token * H_DIM;
  float acc = 0.f;
  for (int k = 0; k < H_DIM; ++k) acc += hp[k] * gw[k * N_EXP + e];
  lg[tid] = acc;
  logits[(size_t)token * N_EXP + e] = acc;
  __syncthreads();
  if (tid < 32) {
    const int t2 = blockIdx.x * 32 + tid;
    float v[8];
#pragma unroll
    for (int q = 0; q < 8; ++q) v[q] = lg[tid * 8 + q];
    int i0 = 0;
#pragma unroll
    for (int q = 1; q < 8; ++q) if (v[q] > v[i0]) i0 = q;
    int i1 = (i0 == 0) ? 1 : 0;
#pragma unroll
    for (int q = 0; q < 8; ++q) if (q != i0 && v[q] > v[i1]) i1 = q;
    const float eb = __expf(v[i1] - v[i0]);
    const float w0 = 1.0f / (1.0f + eb);
    const float w1v = eb / (1.0f + eb);
#pragma unroll
    for (int q = 0; q < 8; ++q)
      wbuf[(size_t)t2 * N_EXP + q] = (q == i0) ? w0 : ((q == i1) ? w1v : 0.0f);
  }
}

// ------------- Stable per-expert token compaction (1 wave per expert, deterministic) --
__global__ void moe_gather_kernel(const float* __restrict__ wbuf,
                                  int* __restrict__ cnt,
                                  int* __restrict__ lists, int T) {
  const int e = blockIdx.x;
  const int lane = threadIdx.x;
  int base = 0;
  for (int t0 = 0; t0 < T; t0 += 32) {
    const int tok = t0 + lane;
    const bool sel = wbuf[(size_t)tok * N_EXP + e] > 0.0f;  // softmax weights > 0
    const unsigned int m = (unsigned int)__ballot(sel);
    const int pre = __popc(m & ((1u << lane) - 1u));
    if (sel) lists[(size_t)e * T + base + pre] = tok;
    base += __popc(m);
  }
  if (lane == 0) cnt[e] = base;
}

// -------- Expert-sparse FFN: one (expert, 32-gathered-token tile) per workgroup -------
__global__ __launch_bounds__(512)
void moe_ffn_kernel(const float* __restrict__ hs,
                    const float* __restrict__ w1,
                    const float* __restrict__ w3,
                    const float* __restrict__ w2,
                    const float* __restrict__ wbuf,
                    const int* __restrict__ cnt,
                    const int* __restrict__ lists,
                    float* __restrict__ out, int T) {
  const int e = blockIdx.y;
  const int cn = cnt[e];
  if ((int)blockIdx.x * M_TILE >= cn) return;  // uniform early exit

  __shared__ unsigned int xa[2 * 32 * 32 * 8];      // 64 KB: x bf16, A-fragment order
  __shared__ unsigned int stg[16 * 4 * SLOT_DW];    // 136 KB: 4 skewed slots / wave
  __shared__ unsigned short tl[M_TILE * TROW];      // 17 KB: silu(xw1)*xw3 chunk, bf16
  __shared__ int toks[M_TILE];
  __shared__ float wrow[M_TILE];

  const int tid   = threadIdx.x;
  const int lane  = tid & 31;
  const int wave  = tid >> 5;       // GEMM1: 16-col tile in chunk; GEMM2: 64-col H slice
  const int lhalf = lane >> 4;
  const int l15   = lane & 15;
  unsigned int* stg_w = stg + wave * (4 * SLOT_DW);

  if (tid < M_TILE) {
    const int idx = blockIdx.x * M_TILE + tid;
    const int g = (idx < cn) ? idx : 0;
    const int tok = lists[(size_t)e * T + g];
    toks[tid] = tok;
    wrow[tid] = (idx < cn) ? wbuf[(size_t)tok * N_EXP + e] : 0.0f;  // pad rows weight 0
  }
  __syncthreads();

  // Stage gathered x rows (fp32 -> bf16) into A-fragment order.
  for (int it = 0; it < 32; ++it) {
    int s  = tid + it * 512;
    int j  = s & 7;
    int ln = (s >> 3) & 31;
    int kb = (s >> 8) & 31;
    int m2 = s >> 13;
    int koff = ((ln >> 4) << 3) + ((j < 4) ? (j << 1) : (16 + ((j - 4) << 1)));
    int k = (kb << 5) + koff;
    const float* src = hs + (size_t)toks[(m2 << 4) + (ln & 15)] * H_DIM + k;
    xa[s] = cvtpk(src[0], src[1]);
  }
  __syncthreads();

  const float* w1e = w1 + (size_t)e * H_DIM * I_DIM;
  const float* w3e = w3 + (size_t)e * H_DIM * I_DIM;
  const float* w2e = w2 + (size_t)e * I_DIM * H_DIM;

  FragAcc acc2[2][4];
#pragma unroll
  for (int m = 0; m < 2; ++m)
#pragma unroll
    for (int n = 0; n < 4; ++n)
#pragma unroll
      for (int r = 0; r < 8; ++r) acc2[m][n].f[r] = 0.f;

  for (int ic = 0; ic < I_DIM; ic += IC) {
    // ---- GEMM1: P = x@w1, Q = x@w3; this wave owns cols [ic+wave*16, +16), both mi ----
    FragAcc P[2], Q[2];
#pragma unroll
    for (int m = 0; m < 2; ++m)
#pragma unroll
      for (int r = 0; r < 8; ++r) { P[m].f[r] = 0.f; Q[m].f[r] = 0.f; }

    const int col1 = ic + (wave << 4);
    stage_tile(w1e + col1, I_DIM, stg_w + 0 * SLOT_DW, lane);
    stage_tile(w3e + col1, I_DIM, stg_w + 2 * SLOT_DW, lane);
#pragma unroll 1
    for (int ks = 0; ks < 32; ++ks) {
      const int cur = ks & 1;
      if (ks + 1 < 32) {
        stage_tile(w1e + (size_t)((ks + 1) << 5) * I_DIM + col1, I_DIM,
                   stg_w + (cur ^ 1) * SLOT_DW, lane);
        stage_tile(w3e + (size_t)((ks + 1) << 5) * I_DIM + col1, I_DIM,
                   stg_w + (2 + (cur ^ 1)) * SLOT_DW, lane);
        wait_async_c<8>();   // 8 newly issued outstanding; current tiles complete
      } else {
        wait_async_c<0>();
      }
      Frag16 b1, b3;
      build_fragB(b1, stg_w + cur * SLOT_DW, lhalf, l15);
      build_fragB(b3, stg_w + (2 + cur) * SLOT_DW, lhalf, l15);
#pragma unroll
      for (int mi = 0; mi < 2; ++mi) {
        Frag16 a;
        const uint4* p4 = (const uint4*)&xa[(((mi * 32 + ks) * 32) + lane) * 8];
        uint4 lo = p4[0], hi = p4[1];
        a.d[0] = lo.x; a.d[1] = lo.y; a.d[2] = lo.z; a.d[3] = lo.w;
        a.d[4] = hi.x; a.d[5] = hi.y; a.d[6] = hi.z; a.d[7] = hi.w;
        P[mi].v = wmma_bf16(a, b1, P[mi].v);
        Q[mi].v = wmma_bf16(a, b3, Q[mi].v);
      }
    }
    __syncthreads();  // prior chunk's t fully consumed (also drains DS before reuse)
    {
      const int colt = (wave << 4) + l15;
#pragma unroll
      for (int mi = 0; mi < 2; ++mi) {
        const int row0 = (mi << 4) + (lhalf << 3);
#pragma unroll
        for (int r = 0; r < 8; ++r) {
          float p = P[mi].f[r];
          float s = p / (1.0f + __expf(-p));  // silu
          tl[(row0 + r) * TROW + colt] = f2bf(s * Q[mi].f[r]);
        }
      }
    }
    __syncthreads();
    // ---- GEMM2: acc2 += t[32,256] @ w2[ic:ic+256, wave*64 : +64], both mi ----
    // 4-slot rotation, two 2KB tiles in flight.
    const unsigned int* t32 = (const unsigned int*)tl;
    const int hbase = wave << 6;
    stage_tile(w2e + (size_t)ic * H_DIM + hbase, H_DIM, stg_w + 0 * SLOT_DW, lane);
    stage_tile(w2e + (size_t)ic * H_DIM + hbase + 16, H_DIM, stg_w + 1 * SLOT_DW, lane);
    int sidx = 0;
#pragma unroll 1
    for (int kb = 0; kb < NKB; ++kb) {
      Frag16 a0, a1;
#pragma unroll
      for (int j = 0; j < 8; ++j) {
        const int koff = (lhalf << 3) + ((j < 4) ? (j << 1) : (16 + ((j - 4) << 1)));
        const int k = (kb << 5) + koff;
        a0.d[j] = t32[l15 * (TROW / 2) + (k >> 1)];
        a1.d[j] = t32[(16 + l15) * (TROW / 2) + (k >> 1)];
      }
#pragma unroll
      for (int nt = 0; nt < 4; ++nt) {
        if (sidx + 2 < NKB * 4) {
          const int nn = sidx + 2;
          stage_tile(w2e + (size_t)(ic + ((nn >> 2) << 5)) * H_DIM + hbase + ((nn & 3) << 4),
                     H_DIM, stg_w + (nn & 3) * SLOT_DW, lane);
          wait_async_c<8>();   // tiles sidx+1, sidx+2 outstanding; tile sidx complete
        } else if (sidx + 1 < NKB * 4) {
          wait_async_c<4>();
        } else {
          wait_async_c<0>();
        }
        Frag16 b;
        build_fragB(b, stg_w + (sidx & 3) * SLOT_DW, lhalf, l15);
        acc2[0][nt].v = wmma_bf16(a0, b, acc2[0][nt].v);
        acc2[1][nt].v = wmma_bf16(a1, b, acc2[1][nt].v);
        ++sidx;
      }
    }
  }  // ic chunks

  // Weighted atomic accumulation: exactly TOP_K commutative f32 adds per element
  // on a zero-initialized output -> bitwise deterministic.
#pragma unroll
  for (int mi = 0; mi < 2; ++mi) {
    const int row0 = (mi << 4) + (lhalf << 3);
#pragma unroll
    for (int r = 0; r < 8; ++r) {
      const float wv = wrow[row0 + r];
      float* orow = out + (size_t)toks[row0 + r] * H_DIM + (wave << 6) + l15;
#pragma unroll
      for (int nt = 0; nt < 4; ++nt)
        unsafeAtomicAdd(orow + (nt << 4), wv * acc2[mi][nt].f[r]);
    }
  }
}

extern "C" void kernel_launch(void* const* d_in, const int* in_sizes, int n_in,
                              void* d_out, int out_size, void* d_ws, size_t ws_size,
                              hipStream_t stream) {
  const float* hs = (const float*)d_in[0];
  const float* gw = (const float*)d_in[1];
  const float* w1 = (const float*)d_in[2];
  const float* w3 = (const float*)d_in[3];
  const float* w2 = (const float*)d_in[4];
  float* out = (float*)d_out;
  const int T = in_sizes[0] / H_DIM;  // B*S tokens
  float* logits = out + (size_t)T * H_DIM;

  float* wbuf = (float*)d_ws;                                      // [T*8] weights
  int* cnt   = (int*)((char*)d_ws + (size_t)T * N_EXP * 4);        // [8]
  int* lists = (int*)((char*)d_ws + (size_t)T * N_EXP * 4 + 256);  // [8*T]

  (void)hipMemsetAsync(out, 0, (size_t)T * H_DIM * sizeof(float), stream);
  moe_router_kernel<<<dim3(T / 32), dim3(256), 0, stream>>>(hs, gw, logits, wbuf);
  moe_gather_kernel<<<dim3(N_EXP), dim3(32), 0, stream>>>(wbuf, cnt, lists, T);
  moe_ffn_kernel<<<dim3(T / M_TILE, N_EXP), dim3(512), 0, stream>>>(
      hs, w1, w3, w2, wbuf, cnt, lists, out, T);
}
